// GATLayer_80659485819333
// MI455X (gfx1250) — matrix operand order repaired and checked by reading, not compiled, not run on hardware
//
#include <hip/hip_runtime.h>
#include <hip/hip_bf16.h>

#define N_NODES   50000
#define N_EDGES   800000
#define E_TOT     (N_EDGES + N_NODES)
#define IN_CH     256
#define HEADS     8
#define OUT_CH    32
#define HC        (HEADS * OUT_CH)   // 256
#define NEG_SLOPE 0.2f
#define EPS_F     1e-16f

typedef __attribute__((ext_vector_type(16))) _Float16 v16h;
typedef __attribute__((ext_vector_type(8)))  float    v8f;

// ---------------------------------------------------------------------------
// Order-preserving float <-> unsigned key mapping for atomicMax on floats.
// key(f) is strictly increasing in f; all finite floats map to key > 0, so a
// zero-initialized key array acts as "-inf" (every dst has a self-loop, so
// every slot is written at least once before being read back).
// ---------------------------------------------------------------------------
__device__ __forceinline__ unsigned fkey(float f) {
    unsigned b = __float_as_uint(f);
    return (b & 0x80000000u) ? ~b : (b | 0x80000000u);
}
__device__ __forceinline__ float funkey(unsigned k) {
    unsigned b = (k & 0x80000000u) ? (k & 0x7FFFFFFFu) : ~k;
    return __uint_as_float(b);
}

// ---------------------------------------------------------------------------
// Kernel 1: Wh = x @ W^T  via v_wmma_f32_16x16x32_f16 (f32 accumulate).
// One wave (32 threads) per 16(M) x 64(N) tile: 4 accumulators, A reused 4x.
// grid = (N_NODES/16, HC/64), block = 32.
//
// Fragment layouts used here (wave32; A and B share contiguous k order):
//   A (16x32 f16): lane = {M = lane&15}; v16h elem i = K = kh*16 + i,
//                  kh = lane>>4.
//   B (32x16 f16): lane = {N = lane&15}; v16h elem i = K = kh*16 + i.
//   C/D (16x16 f32): VGPR r: M = r + kh*8, N = lane&15.
// ---------------------------------------------------------------------------
__global__ void __launch_bounds__(32)
gat_gemm_wmma(const float* __restrict__ x, const float* __restrict__ W,
              float* __restrict__ Wh) {
    const int lane  = threadIdx.x;
    const int mrow  = lane & 15;
    const int khalf = lane >> 4;
    const int m0 = blockIdx.x * 16;
    const int n0 = blockIdx.y * 64;

    const float* xrow  = x + (size_t)(m0 + mrow) * IN_CH;
    const float* wrow0 = W + (size_t)(n0 + mrow) * IN_CH;

    v8f acc0 = {}, acc1 = {}, acc2 = {}, acc3 = {};

    for (int k = 0; k < IN_CH; k += 32) {
        v16h a;
#pragma unroll
        for (int i = 0; i < 8; ++i) {
            a[i]     = (_Float16)xrow[k + khalf * 16 + i];
            a[i + 8] = (_Float16)xrow[k + khalf * 16 + 8 + i];
        }
        const int kb = k + khalf * 16;
        v16h b0, b1, b2, b3;
#pragma unroll
        for (int i = 0; i < 16; ++i) {
            b0[i] = (_Float16)wrow0[0 * 16 * IN_CH + kb + i];
            b1[i] = (_Float16)wrow0[1 * 16 * IN_CH + kb + i];
            b2[i] = (_Float16)wrow0[2 * 16 * IN_CH + kb + i];
            b3[i] = (_Float16)wrow0[3 * 16 * IN_CH + kb + i];
        }
        acc0 = __builtin_amdgcn_wmma_f32_16x16x32_f16(false, a, false, b0, (short)0, acc0, false, false);
        acc1 = __builtin_amdgcn_wmma_f32_16x16x32_f16(false, a, false, b1, (short)0, acc1, false, false);
        acc2 = __builtin_amdgcn_wmma_f32_16x16x32_f16(false, a, false, b2, (short)0, acc2, false, false);
        acc3 = __builtin_amdgcn_wmma_f32_16x16x32_f16(false, a, false, b3, (short)0, acc3, false, false);
    }

#pragma unroll
    for (int r = 0; r < 8; ++r) {
        float* orow = Wh + (size_t)(m0 + r + khalf * 8) * HC + n0 + mrow;
        orow[0]  = acc0[r];
        orow[16] = acc1[r];
        orow[32] = acc2[r];
        orow[48] = acc3[r];
    }
}

// ---------------------------------------------------------------------------
// Kernel 2: per-(node, head) attention halves.
// ---------------------------------------------------------------------------
__global__ void __launch_bounds__(256)
att_halves(const float* __restrict__ Wh, const float* __restrict__ a_left,
           const float* __restrict__ a_right, float* __restrict__ el,
           float* __restrict__ er) {
    int t = blockIdx.x * blockDim.x + threadIdx.x;
    if (t >= N_NODES * HEADS) return;
    int h = t & (HEADS - 1);
    const float* w  = Wh + (size_t)(t >> 3) * HC + h * OUT_CH;
    const float* al = a_left  + h * OUT_CH;
    const float* ar = a_right + h * OUT_CH;
    float sl = 0.f, sr = 0.f;
#pragma unroll
    for (int i = 0; i < OUT_CH; ++i) {
        float v = w[i];
        sl += v * al[i];
        sr += v * ar[i];
    }
    el[t] = sl;
    er[t] = sr;
}

// ---------------------------------------------------------------------------
// Kernel 3: per-(edge, head) raw score + leaky ReLU + segment-max (atomicMax
// on monotonic unsigned key, grouped by dst).
// ---------------------------------------------------------------------------
__global__ void __launch_bounds__(256)
edge_score_max(const int* __restrict__ ei, const float* __restrict__ el,
               const float* __restrict__ er, float* __restrict__ ebuf,
               unsigned* __restrict__ emax) {
    int t = blockIdx.x * blockDim.x + threadIdx.x;
    if (t >= E_TOT * HEADS) return;
    int e = t >> 3, h = t & (HEADS - 1);
    int s, d;
    if (e < N_EDGES) { s = ei[e]; d = ei[N_EDGES + e]; }
    else             { s = e - N_EDGES; d = s; }
    float v = el[s * HEADS + h] + er[d * HEADS + h];
    v = (v >= 0.f) ? v : NEG_SLOPE * v;
    ebuf[t] = v;
    atomicMax(&emax[d * HEADS + h], fkey(v));
}

// ---------------------------------------------------------------------------
// Kernel 4: per-(edge, head) exp(e - max) + segment-sum denominator.
// ---------------------------------------------------------------------------
__global__ void __launch_bounds__(256)
edge_exp_sum(const int* __restrict__ ei, float* __restrict__ ebuf,
             const unsigned* __restrict__ emax, float* __restrict__ denom) {
    int t = blockIdx.x * blockDim.x + threadIdx.x;
    if (t >= E_TOT * HEADS) return;
    int e = t >> 3, h = t & (HEADS - 1);
    int d = (e < N_EDGES) ? ei[N_EDGES + e] : (e - N_EDGES);
    float m  = funkey(emax[d * HEADS + h]);
    float ex = __expf(ebuf[t] - m);
    ebuf[t] = ex;
    atomicAdd(&denom[d * HEADS + h], ex);
}

// ---------------------------------------------------------------------------
// Kernel 5: aggregation. One 256-thread block per edge: coalesced 1KB gather
// of Wh[src], scaled by alpha[h], non-returning f32 atomics into out[dst].
// This pass dominates runtime (~1.7 GB total traffic, mostly L2-served).
// ---------------------------------------------------------------------------
__global__ void __launch_bounds__(256)
edge_aggregate(const int* __restrict__ ei, const float* __restrict__ Wh,
               const float* __restrict__ ebuf, const float* __restrict__ denom,
               float* __restrict__ out) {
    int e = blockIdx.x;
    int t = threadIdx.x;
    int s, d;
    if (e < N_EDGES) { s = ei[e]; d = ei[N_EDGES + e]; }
    else             { s = e - N_EDGES; d = s; }
    int h = t >> 5;  // OUT_CH == 32 elements per head
    float alpha = ebuf[e * HEADS + h] / (denom[d * HEADS + h] + EPS_F);
    float v = Wh[(size_t)s * HC + t] * alpha;
    atomicAdd(&out[(size_t)d * HC + t], v);
}

// ---------------------------------------------------------------------------
extern "C" void kernel_launch(void* const* d_in, const int* in_sizes, int n_in,
                              void* d_out, int out_size, void* d_ws, size_t ws_size,
                              hipStream_t stream) {
    const float* x       = (const float*)d_in[0];
    const int*   ei      = (const int*)d_in[1];
    const float* W       = (const float*)d_in[2];
    const float* a_left  = (const float*)d_in[3];
    const float* a_right = (const float*)d_in[4];
    float*       out     = (float*)d_out;

    // workspace layout (floats): Wh | el | er | ebuf | emax(keys) | denom
    float*    Wh    = (float*)d_ws;
    float*    el    = Wh + (size_t)N_NODES * HC;
    float*    er    = el + (size_t)N_NODES * HEADS;
    float*    ebuf  = er + (size_t)N_NODES * HEADS;
    unsigned* emax  = (unsigned*)(ebuf + (size_t)E_TOT * HEADS);
    float*    denom = (float*)(emax + (size_t)N_NODES * HEADS);

    hipMemsetAsync(out,   0, sizeof(float)    * (size_t)out_size,        stream);
    hipMemsetAsync(emax,  0, sizeof(unsigned) * (size_t)N_NODES * HEADS, stream);
    hipMemsetAsync(denom, 0, sizeof(float)    * (size_t)N_NODES * HEADS, stream);

    dim3 gemm_grid(N_NODES / 16, HC / 64);
    gat_gemm_wmma<<<gemm_grid, 32, 0, stream>>>(x, W, Wh);

    int nh = N_NODES * HEADS;
    att_halves<<<(nh + 255) / 256, 256, 0, stream>>>(Wh, a_left, a_right, el, er);

    int eth = E_TOT * HEADS;
    edge_score_max<<<(eth + 255) / 256, 256, 0, stream>>>(ei, el, er, ebuf, emax);
    edge_exp_sum<<<(eth + 255) / 256, 256, 0, stream>>>(ei, ebuf, emax, denom);

    edge_aggregate<<<E_TOT, 256, 0, stream>>>(ei, Wh, ebuf, denom, out);
}
